// SGEDiff_55070070669491
// MI455X (gfx1250) — matrix operand order repaired and checked by reading, not compiled, not run on hardware
//
#include <hip/hip_runtime.h>
#include <hip/hip_bf16.h>
#include <math.h>

typedef __attribute__((ext_vector_type(16))) _Float16 v16h;
typedef __attribute__((ext_vector_type(8)))  float    v8f;

#define WMMA_F16(a, b, c) \
  __builtin_amdgcn_wmma_f32_16x16x32_f16(false, (a), false, (b), (short)0, (c), false, false)

// ---------------------------------------------------------------------------
// float atomic max via int/uint ordering trick (buffer must be init'd to -FLT_MAX)
// ---------------------------------------------------------------------------
__device__ __forceinline__ void atomicMaxF(float* addr, float val) {
  if (val >= 0.0f) atomicMax((int*)addr, __float_as_int(val));
  else             atomicMin((unsigned int*)addr, (unsigned int)__float_as_int(val));
}

// ---------------------------------------------------------------------------
// Pack a f32 row-major weight slice W[row0 : row0+rows, 0:ncols] into f16 WMMA
// B-fragments. Fragment layout (per 32x16 K-N block): lane l holds column
// n = nb*16 + (l&15), elements i=0..15 hold K = kb*32 + (l>>4)*8 + i + (i<8 ? 0 : 8).
// Storage: out[((kb*nbCount + nb)*32 + lane)*16 + i]
// ---------------------------------------------------------------------------
__global__ void pack_weight_f16(const float* __restrict__ W, int row0, int rows,
                                int ncols, int kbCount, int nbCount,
                                _Float16* __restrict__ out) {
  int idx = blockIdx.x * blockDim.x + threadIdx.x;
  int total = kbCount * nbCount * 32 * 16;
  if (idx >= total) return;
  int i    = idx & 15;
  int lane = (idx >> 4) & 31;
  int blk  = idx >> 9;
  int nb   = blk % nbCount;
  int kb   = blk / nbCount;
  int n = nb * 16 + (lane & 15);
  int k = kb * 32 + ((lane >> 4) << 3) + i + ((i < 8) ? 0 : 8);
  float v = 0.0f;
  if (k < rows && n < ncols) v = W[(size_t)(row0 + k) * ncols + n];
  out[idx] = (_Float16)v;
}

// ---------------------------------------------------------------------------
// generic fill
// ---------------------------------------------------------------------------
__global__ void fillf(float* __restrict__ p, float v, int n) {
  int i = blockIdx.x * blockDim.x + threadIdx.x;
  if (i < n) p[i] = v;
}

// ---------------------------------------------------------------------------
// Per-edge preprocessing: rel_x, dist, gaussian smearing, outer product ->
// geo[E][96] f16 ([edge_attr(4) | r_feat(80) | pad(12)]), plus the two
// sigmoid edge weights (x2h and h2x) and rel_x.
// ---------------------------------------------------------------------------
__global__ void edge_pre(const float* __restrict__ x, const float* __restrict__ ea,
                         const long long* __restrict__ eidx,
                         const float* __restrict__ ewwA, const float* __restrict__ ewbA,
                         const float* __restrict__ ewwB, const float* __restrict__ ewbB,
                         _Float16* __restrict__ geo, float* __restrict__ relx,
                         float* __restrict__ ewA, float* __restrict__ ewB, int E_) {
  int e = blockIdx.x * blockDim.x + threadIdx.x;
  if (e >= E_) return;
  long long s = eidx[e];
  long long d = eidx[E_ + e];
  float rx[3], d2 = 0.f;
#pragma unroll
  for (int i = 0; i < 3; ++i) {
    rx[i] = x[d * 3 + i] - x[s * 3 + i];
    d2 += rx[i] * rx[i];
    relx[(size_t)e * 3 + i] = rx[i];
  }
  float dist = sqrtf(d2);
  const float step  = 10.0f / 19.0f;
  const float coeff = -0.5f / (step * step);
  float g[20];
#pragma unroll
  for (int j = 0; j < 20; ++j) {
    float df = dist - step * (float)j;
    g[j] = __expf(coeff * df * df);
  }
  _Float16* gp = geo + (size_t)e * 96;
  float a4[4];
#pragma unroll
  for (int f = 0; f < 4; ++f) {
    a4[f] = ea[(size_t)e * 4 + f];
    gp[f] = (_Float16)a4[f];
  }
  float dA = 0.f, dB = 0.f;
#pragma unroll
  for (int f = 0; f < 4; ++f)
#pragma unroll
    for (int j = 0; j < 20; ++j) {
      float r = a4[f] * g[j];
      gp[4 + f * 20 + j] = (_Float16)r;
      dA += r * ewwA[f * 20 + j];
      dB += r * ewwB[f * 20 + j];
    }
#pragma unroll
  for (int c = 84; c < 96; ++c) gp[c] = (_Float16)0.0f;
  ewA[e] = 1.0f / (1.0f + __expf(-(dA + ewbA[0])));
  ewB[e] = 1.0f / (1.0f + __expf(-(dB + ewbB[0])));
}

// ---------------------------------------------------------------------------
// Node-level linear: Y[N,dout] = X[N,din] @ Wpacked   (one wave per 16 rows)
// ---------------------------------------------------------------------------
__global__ void node_linear(const float* __restrict__ X, int din,
                            const _Float16* __restrict__ Wp,
                            float* __restrict__ Y, int dout) {
  int lane = threadIdx.x;
  int row0 = blockIdx.x * 16;
  int m = lane & 15, klo = (lane >> 4) * 8, half = lane >> 4, colL = lane & 15;
  int kbcnt = din >> 5, nbcnt = dout >> 4;
  const float* xr = X + (size_t)(row0 + m) * din;
  for (int nb = 0; nb < nbcnt; ++nb) {
    v8f acc = {};
    for (int kb = 0; kb < kbcnt; ++kb) {
      v16h a;
#pragma unroll
      for (int i = 0; i < 16; ++i) {
        int k = kb * 32 + klo + i + ((i < 8) ? 0 : 8);
        a[i] = (_Float16)xr[k];
      }
      v16h b = *(const v16h*)(Wp + (((size_t)(kb * nbcnt + nb) * 32 + lane) << 4));
      acc = WMMA_F16(a, b, acc);
    }
#pragma unroll
    for (int j = 0; j < 8; ++j)
      Y[(size_t)(row0 + half * 8 + j) * dout + nb * 16 + colL] = acc[j];
  }
}

// ---------------------------------------------------------------------------
// Node-level full MLP: Y = relu(LN(X@W1 + b1))@W2 + b2 (+res), dout fixed 128
// ---------------------------------------------------------------------------
__global__ void node_mlp(const float* __restrict__ X, int din,
                         const _Float16* __restrict__ W1p, const float* __restrict__ b1,
                         const float* __restrict__ lng, const float* __restrict__ lnb,
                         const _Float16* __restrict__ W2p, const float* __restrict__ b2,
                         const float* __restrict__ res, float* __restrict__ Y) {
  __shared__ float S[16][128];
  __shared__ float mean[16], rstd[16];
  int lane = threadIdx.x;
  int row0 = blockIdx.x * 16;
  int m = lane & 15, klo = (lane >> 4) * 8, half = lane >> 4, colL = lane & 15;
  int kbcnt = din >> 5;
  const float* xr = X + (size_t)(row0 + m) * din;
  // layer 1
  for (int nb = 0; nb < 8; ++nb) {
    v8f acc = {};
    for (int kb = 0; kb < kbcnt; ++kb) {
      v16h a;
#pragma unroll
      for (int i = 0; i < 16; ++i) {
        int k = kb * 32 + klo + i + ((i < 8) ? 0 : 8);
        a[i] = (_Float16)xr[k];
      }
      v16h b = *(const v16h*)(W1p + (((size_t)(kb * 8 + nb) * 32 + lane) << 4));
      acc = WMMA_F16(a, b, acc);
    }
#pragma unroll
    for (int j = 0; j < 8; ++j)
      S[half * 8 + j][nb * 16 + colL] = acc[j] + b1[nb * 16 + colL];
  }
  __syncthreads();
  // LayerNorm stats (2 lanes per row)
  {
    int r = lane >> 1, p = lane & 1;
    float s1 = 0.f, s2 = 0.f;
    for (int c = p * 64; c < p * 64 + 64; ++c) { float v = S[r][c]; s1 += v; s2 += v * v; }
    s1 += __shfl_xor(s1, 1);
    s2 += __shfl_xor(s2, 1);
    float mu = s1 * (1.0f / 128.0f);
    float var = s2 * (1.0f / 128.0f) - mu * mu;
    if (p == 0) { mean[r] = mu; rstd[r] = rsqrtf(var + 1e-5f); }
  }
  __syncthreads();
  // layer 2
  for (int nb = 0; nb < 8; ++nb) {
    v8f acc = {};
    for (int kb = 0; kb < 4; ++kb) {
      v16h a;
      float mu = mean[m], rs = rstd[m];
#pragma unroll
      for (int i = 0; i < 16; ++i) {
        int k = kb * 32 + klo + i + ((i < 8) ? 0 : 8);
        float v = (S[m][k] - mu) * rs * lng[k] + lnb[k];
        v = v > 0.f ? v : 0.f;
        a[i] = (_Float16)v;
      }
      v16h b = *(const v16h*)(W2p + (((size_t)(kb * 8 + nb) * 32 + lane) << 4));
      acc = WMMA_F16(a, b, acc);
    }
#pragma unroll
    for (int j = 0; j < 8; ++j) {
      size_t oi = (size_t)(row0 + half * 8 + j) * 128 + nb * 16 + colL;
      float v = acc[j] + b2[nb * 16 + colL];
      if (res) v += res[oi];
      Y[oi] = v;
    }
  }
}

// ---------------------------------------------------------------------------
// Edge MLP helper: S <- PA[a]+PB[b]+b1 ; S += geo@W1geo ; LN ; O <- relu@W2+b2
// ---------------------------------------------------------------------------
__device__ __forceinline__ void run_edge_mlp(
    int lane, int e0, const _Float16* __restrict__ geo,
    const int* An, const int* Bn,
    const float* __restrict__ PA, const float* __restrict__ PB,
    const float* __restrict__ b1,
    const float* __restrict__ lng, const float* __restrict__ lnb,
    const _Float16* __restrict__ W1g, const _Float16* __restrict__ W2,
    const float* __restrict__ b2, int nbcnt,
    float (*S)[128], float (*O)[128], float* mean, float* rstd) {
  int m = lane & 15, klo = (lane >> 4) * 8, half = lane >> 4, colL = lane & 15;
  // gather part of layer 1 (+ bias)
  for (int t = 0; t < 64; ++t) {
    int idx = lane * 64 + t;
    int r = idx >> 7, c = idx & 127;
    S[r][c] = PA[(size_t)An[r] * 128 + c] + PB[(size_t)Bn[r] * 128 + c] + b1[c];
  }
  __syncthreads();
  // geo part of layer 1 (K = 96, 3 k-blocks)
  const _Float16* grow = geo + (size_t)(e0 + m) * 96;
  for (int nb = 0; nb < 8; ++nb) {
    v8f acc = {};
    for (int kb = 0; kb < 3; ++kb) {
      v16h a;
#pragma unroll
      for (int i = 0; i < 16; ++i) {
        int k = kb * 32 + klo + i + ((i < 8) ? 0 : 8);
        a[i] = grow[k];
      }
      v16h b = *(const v16h*)(W1g + (((size_t)(kb * 8 + nb) * 32 + lane) << 4));
      acc = WMMA_F16(a, b, acc);
    }
#pragma unroll
    for (int j = 0; j < 8; ++j) S[half * 8 + j][nb * 16 + colL] += acc[j];
  }
  __syncthreads();
  // LayerNorm stats
  {
    int r = lane >> 1, p = lane & 1;
    float s1 = 0.f, s2 = 0.f;
    for (int c = p * 64; c < p * 64 + 64; ++c) { float v = S[r][c]; s1 += v; s2 += v * v; }
    s1 += __shfl_xor(s1, 1);
    s2 += __shfl_xor(s2, 1);
    float mu = s1 * (1.0f / 128.0f);
    float var = s2 * (1.0f / 128.0f) - mu * mu;
    if (p == 0) { mean[r] = mu; rstd[r] = rsqrtf(var + 1e-5f); }
  }
  __syncthreads();
  // layer 2
  for (int nb = 0; nb < nbcnt; ++nb) {
    v8f acc = {};
    for (int kb = 0; kb < 4; ++kb) {
      v16h a;
      float mu = mean[m], rs = rstd[m];
#pragma unroll
      for (int i = 0; i < 16; ++i) {
        int k = kb * 32 + klo + i + ((i < 8) ? 0 : 8);
        float v = (S[m][k] - mu) * rs * lng[k] + lnb[k];
        v = v > 0.f ? v : 0.f;
        a[i] = (_Float16)v;
      }
      v16h b = *(const v16h*)(W2 + (((size_t)(kb * nbcnt + nb) * 32 + lane) << 4));
      acc = WMMA_F16(a, b, acc);
    }
#pragma unroll
    for (int j = 0; j < 8; ++j)
      O[half * 8 + j][nb * 16 + colL] = acc[j] + b2[nb * 16 + colL];
  }
  __syncthreads();
}

// ---------------------------------------------------------------------------
// Edge pass A: K-MLP + logits + atomicMax(node), then V-MLP (*e_w) stored out.
// One wave per 16 edges.
// ---------------------------------------------------------------------------
__global__ void edge_passA(
    const _Float16* __restrict__ geo, const long long* __restrict__ eidx, int aIsDst,
    const float* __restrict__ PAk, const float* __restrict__ PBk,
    const float* __restrict__ b1k, const float* __restrict__ lngk,
    const float* __restrict__ lnbk, const _Float16* __restrict__ W1gk,
    const _Float16* __restrict__ W2k, const float* __restrict__ b2k,
    const float* __restrict__ PAv, const float* __restrict__ PBv,
    const float* __restrict__ b1v, const float* __restrict__ lngv,
    const float* __restrict__ lnbv, const _Float16* __restrict__ W1gv,
    const _Float16* __restrict__ W2v, const float* __restrict__ b2v, int doutv,
    const float* __restrict__ q, const float* __restrict__ ew,
    float* __restrict__ logits, float* __restrict__ nodeMax,
    _Float16* __restrict__ vout128, float* __restrict__ vout16, int E_) {
  __shared__ float S[16][128];
  __shared__ float O[16][128];
  __shared__ float Q[16][128];
  __shared__ int ldsDst[16], ldsA[16], ldsB[16];
  __shared__ float ldsEw[16], mean[16], rstd[16];

  int lane = threadIdx.x;
  int e0 = blockIdx.x * 16;
  if (lane < 16) {
    int e = e0 + lane;
    int s = (int)eidx[e];
    int d = (int)eidx[E_ + e];
    ldsDst[lane] = d;
    ldsA[lane] = aIsDst ? d : s;
    ldsB[lane] = aIsDst ? s : d;
    ldsEw[lane] = ew[e];
  }
  __syncthreads();
  // gather q[dst] into LDS
  for (int t = 0; t < 64; ++t) {
    int idx = lane * 64 + t;
    int r = idx >> 7, c = idx & 127;
    Q[r][c] = q[(size_t)ldsDst[r] * 128 + c];
  }
  __syncthreads();

  // ---- K MLP ----
  run_edge_mlp(lane, e0, geo, ldsA, ldsB, PAk, PBk, b1k, lngk, lnbk,
               W1gk, W2k, b2k, 8, S, O, mean, rstd);
  // ---- logits + atomicMax ----
#pragma unroll
  for (int t = 0; t < 8; ++t) {
    int p = lane * 8 + t;
    int r = p >> 4, hh = p & 15;
    float dot = 0.f;
#pragma unroll
    for (int d = 0; d < 8; ++d) dot += O[r][hh * 8 + d] * Q[r][hh * 8 + d];
    float lg = dot * 0.35355339059327373f;  // 1/sqrt(8)
    logits[(size_t)(e0 + r) * 16 + hh] = lg;
    atomicMaxF(&nodeMax[(size_t)ldsDst[r] * 16 + hh], lg);
  }
  __syncthreads();

  // ---- V MLP ----
  run_edge_mlp(lane, e0, geo, ldsA, ldsB, PAv, PBv, b1v, lngv, lnbv,
               W1gv, W2v, b2v, doutv >> 4, S, O, mean, rstd);
  if (doutv == 128) {
    for (int t = 0; t < 64; ++t) {
      int idx = lane * 64 + t;
      int r = idx >> 7, c = idx & 127;
      vout128[(size_t)(e0 + r) * 128 + c] = (_Float16)(O[r][c] * ldsEw[r]);
    }
  } else {  // doutv == 16
#pragma unroll
    for (int t = 0; t < 8; ++t) {
      int p = lane * 8 + t;
      int r = p >> 4, c = p & 15;
      vout16[(size_t)(e0 + r) * 16 + c] = O[r][c] * ldsEw[r];
    }
  }
}

// ---------------------------------------------------------------------------
// Softmax pass B (x2h): ex = exp(logit - max); den += ex; num += ex*v
// ---------------------------------------------------------------------------
__global__ void x2h_passB(const float* __restrict__ logits,
                          const float* __restrict__ nodeMax,
                          const long long* __restrict__ eidx,
                          const _Float16* __restrict__ vbuf,
                          float* __restrict__ num, float* __restrict__ den, int E_) {
  int idx = blockIdx.x * blockDim.x + threadIdx.x;
  if (idx >= E_ * 16) return;
  int e = idx >> 4, hh = idx & 15;
  int dn = (int)eidx[E_ + e];
  float ex = __expf(logits[idx] - nodeMax[(size_t)dn * 16 + hh]);
  atomicAdd(&den[(size_t)dn * 16 + hh], ex);
  const _Float16* vp = vbuf + (size_t)e * 128 + hh * 8;
#pragma unroll
  for (int d = 0; d < 8; ++d)
    atomicAdd(&num[(size_t)dn * 128 + hh * 8 + d], ex * (float)vp[d]);
}

// ---------------------------------------------------------------------------
// Softmax pass B (h2x): den += ex; numx[n,h,:] += ex*v*rel_x
// ---------------------------------------------------------------------------
__global__ void h2x_passB(const float* __restrict__ logits,
                          const float* __restrict__ nodeMax,
                          const long long* __restrict__ eidx,
                          const float* __restrict__ v16buf,
                          const float* __restrict__ relx,
                          float* __restrict__ numx, float* __restrict__ den, int E_) {
  int idx = blockIdx.x * blockDim.x + threadIdx.x;
  if (idx >= E_ * 16) return;
  int e = idx >> 4, hh = idx & 15;
  int dn = (int)eidx[E_ + e];
  float ex = __expf(logits[idx] - nodeMax[(size_t)dn * 16 + hh]);
  atomicAdd(&den[(size_t)dn * 16 + hh], ex);
  float m = ex * v16buf[idx];
#pragma unroll
  for (int d = 0; d < 3; ++d)
    atomicAdd(&numx[((size_t)dn * 16 + hh) * 3 + d], m * relx[(size_t)e * 3 + d]);
}

// ---------------------------------------------------------------------------
// cat[n] = [ num/den | h ]   (input for the out-MLP)
// ---------------------------------------------------------------------------
__global__ void finalize_cat(const float* __restrict__ num, const float* __restrict__ den,
                             const float* __restrict__ h, float* __restrict__ cat, int N_) {
  int idx = blockIdx.x * blockDim.x + threadIdx.x;
  if (idx >= N_ * 128) return;
  int n = idx >> 7, c = idx & 127;
  float a = num[idx] / (den[(size_t)n * 16 + (c >> 3)] + 1e-16f);
  cat[(size_t)n * 256 + c] = a;
  cat[(size_t)n * 256 + 128 + c] = h[idx];
}

// ---------------------------------------------------------------------------
// x_new = x + mean_h(numx/den) * mask
// ---------------------------------------------------------------------------
__global__ void finalize_x(const float* __restrict__ numx, const float* __restrict__ den,
                           const float* __restrict__ x, const int* __restrict__ mask,
                           float* __restrict__ out, int N_) {
  int n = blockIdx.x * blockDim.x + threadIdx.x;
  if (n >= N_) return;
  float ml = (float)mask[n];
#pragma unroll
  for (int d = 0; d < 3; ++d) {
    float s = 0.f;
    for (int hh = 0; hh < 16; ++hh)
      s += numx[((size_t)n * 16 + hh) * 3 + d] / (den[(size_t)n * 16 + hh] + 1e-16f);
    out[(size_t)N_ * 128 + (size_t)n * 3 + d] = x[(size_t)n * 3 + d] + (s * (1.0f / 16.0f)) * ml;
  }
}

// ===========================================================================
extern "C" void kernel_launch(void* const* d_in, const int* in_sizes, int n_in,
                              void* d_out, int out_size, void* d_ws, size_t ws_size,
                              hipStream_t stream) {
  (void)n_in; (void)out_size; (void)ws_size;
  const float*     h    = (const float*)d_in[0];
  const float*     x    = (const float*)d_in[1];
  const float*     ea   = (const float*)d_in[2];
  const long long* eidx = (const long long*)d_in[3];
  const int*       mask = (const int*)d_in[4];
#define IN(i) ((const float*)d_in[(i)])
  const float *hk_w1 = IN(5),  *hk_b1 = IN(6),  *hk_g = IN(7),  *hk_be = IN(8),  *hk_w2 = IN(9),  *hk_b2 = IN(10);
  const float *hv_w1 = IN(11), *hv_b1 = IN(12), *hv_g = IN(13), *hv_be = IN(14), *hv_w2 = IN(15), *hv_b2 = IN(16);
  const float *hq_w1 = IN(17), *hq_b1 = IN(18), *hq_g = IN(19), *hq_be = IN(20), *hq_w2 = IN(21), *hq_b2 = IN(22);
  const float *ewwA  = IN(23), *ewbA  = IN(24);
  const float *o_w1  = IN(25), *o_b1  = IN(26), *o_g  = IN(27), *o_be  = IN(28), *o_w2  = IN(29), *o_b2  = IN(30);
  const float *xk_w1 = IN(31), *xk_b1 = IN(32), *xk_g = IN(33), *xk_be = IN(34), *xk_w2 = IN(35), *xk_b2 = IN(36);
  const float *xv_w1 = IN(37), *xv_b1 = IN(38), *xv_g = IN(39), *xv_be = IN(40), *xv_w2 = IN(41), *xv_b2 = IN(42);
  const float *xq_w1 = IN(43), *xq_b1 = IN(44), *xq_g = IN(45), *xq_be = IN(46), *xq_w2 = IN(47), *xq_b2 = IN(48);
  const float *ewwB  = IN(49), *ewbB  = IN(50);
#undef IN

  const int Nn = in_sizes[0] / 128;
  const int Ee = in_sizes[2] / 4;

  // -------- workspace bump allocator --------
  char* wsp = (char*)d_ws;
  size_t off = 0;
  auto alloc = [&](size_t bytes) -> void* {
    void* p = wsp + off;
    off += (bytes + 255) & ~(size_t)255;
    return p;
  };
  // packed weights (f16)
  const size_t FR = 512 * sizeof(_Float16);  // one 32x16 fragment block
  _Float16* pk_hk_g = (_Float16*)alloc(3 * 8 * FR);
  _Float16* pk_hk_A = (_Float16*)alloc(4 * 8 * FR);
  _Float16* pk_hk_B = (_Float16*)alloc(4 * 8 * FR);
  _Float16* pk_hk_2 = (_Float16*)alloc(4 * 8 * FR);
  _Float16* pk_hv_g = (_Float16*)alloc(3 * 8 * FR);
  _Float16* pk_hv_A = (_Float16*)alloc(4 * 8 * FR);
  _Float16* pk_hv_B = (_Float16*)alloc(4 * 8 * FR);
  _Float16* pk_hv_2 = (_Float16*)alloc(4 * 8 * FR);
  _Float16* pk_xk_g = (_Float16*)alloc(3 * 8 * FR);
  _Float16* pk_xk_A = (_Float16*)alloc(4 * 8 * FR);
  _Float16* pk_xk_B = (_Float16*)alloc(4 * 8 * FR);
  _Float16* pk_xk_2 = (_Float16*)alloc(4 * 8 * FR);
  _Float16* pk_xv_g = (_Float16*)alloc(3 * 8 * FR);
  _Float16* pk_xv_A = (_Float16*)alloc(4 * 8 * FR);
  _Float16* pk_xv_B = (_Float16*)alloc(4 * 8 * FR);
  _Float16* pk_xv_2 = (_Float16*)alloc(4 * 1 * FR);
  _Float16* pk_hq_1 = (_Float16*)alloc(4 * 8 * FR);
  _Float16* pk_hq_2 = (_Float16*)alloc(4 * 8 * FR);
  _Float16* pk_xq_1 = (_Float16*)alloc(4 * 8 * FR);
  _Float16* pk_xq_2 = (_Float16*)alloc(4 * 8 * FR);
  _Float16* pk_o_1  = (_Float16*)alloc(8 * 8 * FR);
  _Float16* pk_o_2  = (_Float16*)alloc(4 * 8 * FR);
  // large buffers
  _Float16* geo    = (_Float16*)alloc((size_t)Ee * 96 * sizeof(_Float16));
  float* relx      = (float*)alloc((size_t)Ee * 3 * sizeof(float));
  float* ewA       = (float*)alloc((size_t)Ee * sizeof(float));
  float* ewB       = (float*)alloc((size_t)Ee * sizeof(float));
  float* logits    = (float*)alloc((size_t)Ee * 16 * sizeof(float));
  _Float16* vbuf   = (_Float16*)alloc((size_t)Ee * 128 * sizeof(_Float16));
  float* v16buf    = (float*)alloc((size_t)Ee * 16 * sizeof(float));
  // node buffers
  float* PAk   = (float*)alloc((size_t)Nn * 128 * sizeof(float));
  float* PBk   = (float*)alloc((size_t)Nn * 128 * sizeof(float));
  float* PAv   = (float*)alloc((size_t)Nn * 128 * sizeof(float));
  float* PBv   = (float*)alloc((size_t)Nn * 128 * sizeof(float));
  float* qbuf  = (float*)alloc((size_t)Nn * 128 * sizeof(float));
  float* num   = (float*)alloc((size_t)Nn * 128 * sizeof(float));
  float* den   = (float*)alloc((size_t)Nn * 16 * sizeof(float));
  float* nmax  = (float*)alloc((size_t)Nn * 16 * sizeof(float));
  float* cat   = (float*)alloc((size_t)Nn * 256 * sizeof(float));
  float* numx  = (float*)alloc((size_t)Nn * 48 * sizeof(float));

  auto pack = [&](const float* W, int row0, int rows, int ncols, int kb, int nb, _Float16* o) {
    int total = kb * nb * 512;
    pack_weight_f16<<<dim3((total + 255) / 256), dim3(256), 0, stream>>>(W, row0, rows, ncols, kb, nb, o);
  };
  // ---- pack all weights ----
  pack(hk_w1, 0, 84, 128, 3, 8, pk_hk_g);  pack(hk_w1, 84, 128, 128, 4, 8, pk_hk_A);
  pack(hk_w1, 212, 128, 128, 4, 8, pk_hk_B); pack(hk_w2, 0, 128, 128, 4, 8, pk_hk_2);
  pack(hv_w1, 0, 84, 128, 3, 8, pk_hv_g);  pack(hv_w1, 84, 128, 128, 4, 8, pk_hv_A);
  pack(hv_w1, 212, 128, 128, 4, 8, pk_hv_B); pack(hv_w2, 0, 128, 128, 4, 8, pk_hv_2);
  pack(xk_w1, 0, 84, 128, 3, 8, pk_xk_g);  pack(xk_w1, 84, 128, 128, 4, 8, pk_xk_A);
  pack(xk_w1, 212, 128, 128, 4, 8, pk_xk_B); pack(xk_w2, 0, 128, 128, 4, 8, pk_xk_2);
  pack(xv_w1, 0, 84, 128, 3, 8, pk_xv_g);  pack(xv_w1, 84, 128, 128, 4, 8, pk_xv_A);
  pack(xv_w1, 212, 128, 128, 4, 8, pk_xv_B); pack(xv_w2, 0, 128, 16, 4, 1, pk_xv_2);
  pack(hq_w1, 0, 128, 128, 4, 8, pk_hq_1); pack(hq_w2, 0, 128, 128, 4, 8, pk_hq_2);
  pack(xq_w1, 0, 128, 128, 4, 8, pk_xq_1); pack(xq_w2, 0, 128, 128, 4, 8, pk_xq_2);
  pack(o_w1, 0, 256, 128, 8, 8, pk_o_1);   pack(o_w2, 0, 128, 128, 4, 8, pk_o_2);

  // ---- edge preprocessing ----
  edge_pre<<<dim3((Ee + 255) / 256), dim3(256), 0, stream>>>(
      x, ea, eidx, ewwA, ewbA, ewwB, ewbB, geo, relx, ewA, ewB, Ee);

  const int nTiles = Nn / 16;
  const int eTiles = Ee / 16;
  const float NEG = -3.402823466e38f;

  // ================= x2h =================
  fillf<<<dim3((Nn * 16 + 255) / 256), dim3(256), 0, stream>>>(nmax, NEG, Nn * 16);
  fillf<<<dim3((Nn * 16 + 255) / 256), dim3(256), 0, stream>>>(den, 0.f, Nn * 16);
  fillf<<<dim3((Nn * 128 + 255) / 256), dim3(256), 0, stream>>>(num, 0.f, Nn * 128);

  node_linear<<<dim3(nTiles), dim3(32), 0, stream>>>(h, 128, pk_hk_A, PAk, 128);
  node_linear<<<dim3(nTiles), dim3(32), 0, stream>>>(h, 128, pk_hk_B, PBk, 128);
  node_linear<<<dim3(nTiles), dim3(32), 0, stream>>>(h, 128, pk_hv_A, PAv, 128);
  node_linear<<<dim3(nTiles), dim3(32), 0, stream>>>(h, 128, pk_hv_B, PBv, 128);
  node_mlp<<<dim3(nTiles), dim3(32), 0, stream>>>(h, 128, pk_hq_1, hq_b1, hq_g, hq_be,
                                                  pk_hq_2, hq_b2, nullptr, qbuf);
  edge_passA<<<dim3(eTiles), dim3(32), 0, stream>>>(
      geo, eidx, /*aIsDst=*/1,
      PAk, PBk, hk_b1, hk_g, hk_be, pk_hk_g, pk_hk_2, hk_b2,
      PAv, PBv, hv_b1, hv_g, hv_be, pk_hv_g, pk_hv_2, hv_b2, /*doutv=*/128,
      qbuf, ewA, logits, nmax, vbuf, nullptr, Ee);
  x2h_passB<<<dim3((Ee * 16 + 255) / 256), dim3(256), 0, stream>>>(
      logits, nmax, eidx, vbuf, num, den, Ee);
  finalize_cat<<<dim3((Nn * 128 + 255) / 256), dim3(256), 0, stream>>>(num, den, h, cat, Nn);
  node_mlp<<<dim3(nTiles), dim3(32), 0, stream>>>(cat, 256, pk_o_1, o_b1, o_g, o_be,
                                                  pk_o_2, o_b2, /*res=*/h, (float*)d_out);

  // ================= h2x =================
  const float* nh = (const float*)d_out;  // new_h = x2h_out
  fillf<<<dim3((Nn * 16 + 255) / 256), dim3(256), 0, stream>>>(nmax, NEG, Nn * 16);
  fillf<<<dim3((Nn * 16 + 255) / 256), dim3(256), 0, stream>>>(den, 0.f, Nn * 16);
  fillf<<<dim3((Nn * 48 + 255) / 256), dim3(256), 0, stream>>>(numx, 0.f, Nn * 48);

  node_linear<<<dim3(nTiles), dim3(32), 0, stream>>>(nh, 128, pk_xk_A, PAk, 128);
  node_linear<<<dim3(nTiles), dim3(32), 0, stream>>>(nh, 128, pk_xk_B, PBk, 128);
  node_linear<<<dim3(nTiles), dim3(32), 0, stream>>>(nh, 128, pk_xv_A, PAv, 128);
  node_linear<<<dim3(nTiles), dim3(32), 0, stream>>>(nh, 128, pk_xv_B, PBv, 128);
  node_mlp<<<dim3(nTiles), dim3(32), 0, stream>>>(nh, 128, pk_xq_1, xq_b1, xq_g, xq_be,
                                                  pk_xq_2, xq_b2, nullptr, qbuf);
  edge_passA<<<dim3(eTiles), dim3(32), 0, stream>>>(
      geo, eidx, /*aIsDst=*/0,  // kv_in = [..., new_h[src], new_h[dst]]
      PAk, PBk, xk_b1, xk_g, xk_be, pk_xk_g, pk_xk_2, xk_b2,
      PAv, PBv, xv_b1, xv_g, xv_be, pk_xv_g, pk_xv_2, xv_b2, /*doutv=*/16,
      qbuf, ewB, logits, nmax, nullptr, v16buf, Ee);
  h2x_passB<<<dim3((Ee * 16 + 255) / 256), dim3(256), 0, stream>>>(
      logits, nmax, eidx, v16buf, relx, numx, den, Ee);
  finalize_x<<<dim3((Nn + 255) / 256), dim3(256), 0, stream>>>(
      numx, den, x, mask, (float*)d_out, Nn);
}
